// GraphDenseAttn_23888608100368
// MI455X (gfx1250) — compile-verified, matrix operand order checked by reading, not run on hardware
//
#include <hip/hip_runtime.h>
#include <math.h>

typedef __bf16 bf16;
typedef __attribute__((ext_vector_type(16))) __bf16 v16bf;
typedef __attribute__((ext_vector_type(8)))  __bf16 v8bf;
typedef __attribute__((ext_vector_type(8)))  float  v8f;
typedef __attribute__((ext_vector_type(4)))  unsigned int v4u;
typedef __attribute__((ext_vector_type(8)))  int v8i;
typedef __attribute__((ext_vector_type(4)))  int v4i;

#define E   256
#define BB  32
#define HH  8
#define DH  32
#define NP  512   // padded per-graph node count (N=504 -> 512)
#define MP  (BB * NP)   // 16384 dense rows

// ---------------------------------------------------------------------------
// WMMA helpers: NT fragment loads (K contiguous in memory for both operands).
// A-matrix 16x32 bf16 layout (ISA 7.12.2): lane m = lane&15, half = lane>>4;
// elements 0..7  = K[half*8 .. half*8+7]
// elements 8..15 = K[16+half*8 .. 16+half*8+7]
// For NT GEMM the B operand (rows = out-col, K contiguous) has the SAME
// per-lane addressing, so one loader serves both.
// ---------------------------------------------------------------------------
__device__ __forceinline__ v16bf ldfrag(const bf16* p) {
  v8bf lo = *reinterpret_cast<const v8bf*>(p);
  v8bf hi = *reinterpret_cast<const v8bf*>(p + 16);
  v16bf r;
#pragma unroll
  for (int i = 0; i < 8; ++i) { r[i] = lo[i]; r[i + 8] = hi[i]; }
  return r;
}

__device__ __forceinline__ v8f wmma_bf16(v16bf a, v16bf b, v8f c) {
  return __builtin_amdgcn_wmma_f32_16x16x32_bf16(false, a, false, b, (short)0, c,
                                                 false, false);
}

// ---------------------------------------------------------------------------
// TDM: 1D contiguous tile DMA, global -> LDS (D# per ISA 08_async_tensor §8).
// data_size = 8 bytes; `bytes` must be a multiple of 8 and <= 64K units.
// This toolchain exposes the 6-arg builtin:
//   (uint32x4 g0, int32x8 g1, int32x4 g2, int32x4 g3, int32x8 g4, i32 cpol)
// ---------------------------------------------------------------------------
#if __has_builtin(__builtin_amdgcn_tensor_load_to_lds)
#define HAVE_TDM 1
__device__ __forceinline__ void tdm_load_1d(unsigned int lds_off,
                                            const void* gptr,
                                            unsigned int bytes) {
  unsigned long long ga = (unsigned long long)gptr;
  unsigned int units = bytes >> 3;                       // 8-byte elements
  v4u g0;
  g0[0] = 1u;                                            // count=1, user D#
  g0[1] = lds_off;                                       // lds_addr
  g0[2] = (unsigned int)(ga & 0xFFFFFFFFu);              // global_addr lo
  g0[3] = (unsigned int)((ga >> 32) & 0x1FFFFFFu)        // global_addr hi
        | 0x80000000u;                                   // type = 2 ("image")
  v8i g1;
  g1[0] = (int)(3u << 16);                               // data_size = 8B
  g1[1] = (int)((units & 0xFFFFu) << 16);                // tensor_dim0[15:0]
  g1[2] = (int)((units >> 16) & 0xFFFFu)                 // tensor_dim0[31:16]
        | (int)(1u << 16);                               // tensor_dim1 = 1
  g1[3] = (int)((units & 0xFFFFu) << 16);                // tile_dim0 = units
  g1[4] = 1;                                             // tile_dim1 = 1
  g1[5] = (int)units;                                    // dim0_stride = units
  g1[6] = 0;
  g1[7] = 0;
  v4i z4 = {0, 0, 0, 0};
  v8i z8 = {0, 0, 0, 0, 0, 0, 0, 0};
  __builtin_amdgcn_tensor_load_to_lds(g0, g1, z4, z4, z8, 0);
}
#else
#define HAVE_TDM 0
#endif

// 32x32 output tile per wave (2x2 fragments), K-loop in steps of 32.
__device__ __forceinline__ void gemm_tile32(const bf16* A, int lda,
                                            const bf16* W, int ldb,
                                            int m0, int n0, int K, v8f c[2][2]) {
  const int lane = threadIdx.x & 31;
  const int r  = lane & 15;
  const int h8 = (lane >> 4) * 8;
  for (int k = 0; k < K; k += 32) {
    __builtin_prefetch(A + (size_t)(m0 + r) * lda + k + 64, 0, 1);
    v16bf a0 = ldfrag(A + (size_t)(m0 + r)      * lda + k + h8);
    v16bf a1 = ldfrag(A + (size_t)(m0 + 16 + r) * lda + k + h8);
    v16bf b0 = ldfrag(W + (size_t)(n0 + r)      * ldb + k + h8);
    v16bf b1 = ldfrag(W + (size_t)(n0 + 16 + r) * ldb + k + h8);
    c[0][0] = wmma_bf16(a0, b0, c[0][0]);
    c[0][1] = wmma_bf16(a0, b1, c[0][1]);
    c[1][0] = wmma_bf16(a1, b0, c[1][0]);
    c[1][1] = wmma_bf16(a1, b1, c[1][1]);
  }
}

// ---------------------------------------------------------------------------
// small utility kernels
// ---------------------------------------------------------------------------
__global__ void k_zero(unsigned int* p, long n) {
  long i = blockIdx.x * (long)blockDim.x + threadIdx.x;
  if (i < n) p[i] = 0u;
}

__global__ void k_f2bf(const float* in, bf16* out, long n) {
  long i = blockIdx.x * (long)blockDim.x + threadIdx.x;
  if (i < n) out[i] = (bf16)in[i];
}

__global__ void k_starts(const int* bids, int* starts, int* counts, int T) {
  int i = blockIdx.x * blockDim.x + threadIdx.x;
  if (i >= T) return;
  int b = bids[i];
  atomicAdd(&counts[b], 1);
  if (i == 0 || bids[i - 1] != b) starts[b] = i;
}

// LayerNorm each token row, scatter into zero-padded dense [B, NP, E] as bf16.
__global__ void k_ln_scatter(const float* x, const int* bids, const int* starts,
                             const float* g, const float* be, bf16* denseX) {
  __shared__ float r1[256];
  __shared__ float r2[256];
  int t = blockIdx.x;
  int c = threadIdx.x;
  float v = x[(size_t)t * E + c];
  r1[c] = v; r2[c] = v * v;
  __syncthreads();
  for (int s = 128; s > 0; s >>= 1) {
    if (c < s) { r1[c] += r1[c + s]; r2[c] += r2[c + s]; }
    __syncthreads();
  }
  float mean = r1[0] * (1.f / 256.f);
  float var  = r2[0] * (1.f / 256.f) - mean * mean;
  float inv  = rsqrtf(var + 1e-5f);
  int b = bids[t];
  int pos = t - starts[b];
  float xn = (v - mean) * inv * g[c] + be[c];
  denseX[((size_t)b * NP + pos) * E + c] = (bf16)xn;
}

// ---------------------------------------------------------------------------
// GEMM kernels (block = 256 threads = 8 waves, block tile 64M x 128N)
// ---------------------------------------------------------------------------
__global__ void k_gemm_qkv(const bf16* A, const bf16* W, const float* bqkv,
                           bf16* Qb, bf16* Kb, bf16* VT) {
  v8f c[2][2] = {};
  int w  = threadIdx.x >> 5;
  int m0 = blockIdx.y * 64  + (w & 1) * 32;
  int n0 = blockIdx.x * 128 + (w >> 1) * 32;
  gemm_tile32(A, E, W, E, m0, n0, E, c);
  int lane = threadIdx.x & 31;
  int col0 = lane & 15, mb = (lane >> 4) * 8;
#pragma unroll
  for (int tm = 0; tm < 2; ++tm)
#pragma unroll
    for (int tn = 0; tn < 2; ++tn)
#pragma unroll
      for (int i = 0; i < 8; ++i) {
        int m = m0 + tm * 16 + mb + i;
        int j = n0 + tn * 16 + col0;
        float v = c[tm][tn][i] + bqkv[j];
        int b = m >> 9, nd = m & (NP - 1);
        if (j < 256) {
          int hh = j >> 5, dd = j & 31;
          Qb[((size_t)(b * HH + hh) * NP + nd) * DH + dd] = (bf16)v;
        } else if (j < 512) {
          int jj = j - 256, hh = jj >> 5, dd = jj & 31;
          Kb[((size_t)(b * HH + hh) * NP + nd) * DH + dd] = (bf16)v;
        } else {          // V stored transposed: [B, H, d, NP]  (NT for PV)
          int jj = j - 512, hh = jj >> 5, dd = jj & 31;
          VT[((size_t)(b * HH + hh) * DH + dd) * NP + nd] = (bf16)v;
        }
      }
}

__global__ void k_gemm_oproj(const bf16* A, const bf16* W, const float* bo,
                             const float* x, const int* starts, const int* counts,
                             float* y) {
  v8f c[2][2] = {};
  int w  = threadIdx.x >> 5;
  int m0 = blockIdx.y * 64  + (w & 1) * 32;
  int n0 = blockIdx.x * 128 + (w >> 1) * 32;
  gemm_tile32(A, E, W, E, m0, n0, E, c);
  int lane = threadIdx.x & 31;
  int col0 = lane & 15, mb = (lane >> 4) * 8;
#pragma unroll
  for (int tm = 0; tm < 2; ++tm)
#pragma unroll
    for (int tn = 0; tn < 2; ++tn)
#pragma unroll
      for (int i = 0; i < 8; ++i) {
        int m = m0 + tm * 16 + mb + i;
        int j = n0 + tn * 16 + col0;
        int b = m >> 9, nd = m & (NP - 1);
        if (nd < counts[b]) {   // gather real nodes + residual
          int t = starts[b] + nd;
          y[(size_t)t * E + j] = c[tm][tn][i] + bo[j] + x[(size_t)t * E + j];
        }
      }
}

__global__ void k_gemm_gelu(const bf16* A, const bf16* W, const float* bb,
                            bf16* out) {
  v8f c[2][2] = {};
  int w  = threadIdx.x >> 5;
  int m0 = blockIdx.y * 64  + (w & 1) * 32;
  int n0 = blockIdx.x * 128 + (w >> 1) * 32;
  gemm_tile32(A, E, W, E, m0, n0, E, c);
  int lane = threadIdx.x & 31;
  int col0 = lane & 15, mb = (lane >> 4) * 8;
#pragma unroll
  for (int tm = 0; tm < 2; ++tm)
#pragma unroll
    for (int tn = 0; tn < 2; ++tn)
#pragma unroll
      for (int i = 0; i < 8; ++i) {
        int m = m0 + tm * 16 + mb + i;
        int j = n0 + tn * 16 + col0;
        float v = c[tm][tn][i] + bb[j];
        float g = 0.5f * v * (1.f + erff(v * 0.70710678f));   // exact GELU
        out[(size_t)m * E + j] = (bf16)g;
      }
}

__global__ void k_gemm_f32(const bf16* A, const bf16* W, const float* bb,
                           float* out) {
  v8f c[2][2] = {};
  int w  = threadIdx.x >> 5;
  int m0 = blockIdx.y * 64  + (w & 1) * 32;
  int n0 = blockIdx.x * 128 + (w >> 1) * 32;
  gemm_tile32(A, E, W, E, m0, n0, E, c);
  int lane = threadIdx.x & 31;
  int col0 = lane & 15, mb = (lane >> 4) * 8;
#pragma unroll
  for (int tm = 0; tm < 2; ++tm)
#pragma unroll
    for (int tn = 0; tn < 2; ++tn)
#pragma unroll
      for (int i = 0; i < 8; ++i) {
        int m = m0 + tm * 16 + mb + i;
        int j = n0 + tn * 16 + col0;
        out[(size_t)m * E + j] = c[tm][tn][i] + bb[j];
      }
}

// ---------------------------------------------------------------------------
// Fused flash attention per (b, h): scores = QK^T/sqrt(d) + bias, key-padding
// mask, online softmax, ctx = P V.  attn_bias (260 MB) streamed exactly once.
// K and V^T for this (b,h) are contiguous 32 KB blocks: wave 0 DMAs them into
// LDS via the Tensor Data Mover, so the 8 waves share one copy instead of
// redundantly streaming K/V through the vector pipe.
// ---------------------------------------------------------------------------
__global__ void k_flash(const bf16* Qb, const bf16* Kb, const bf16* VT,
                        const float* bias, const int* counts, bf16* ctxD, int N) {
  __shared__ __align__(16) bf16 ldsK[NP * DH];      // 32 KB
  __shared__ __align__(16) bf16 ldsV[DH * NP];      // 32 KB
  __shared__ __align__(16) bf16 ldsP[8][16 * 32];   // 8 KB
  int bh = blockIdx.x;                // 0 .. B*H-1
  int b  = bh / HH;
  int hh = bh - b * HH;
  int w    = threadIdx.x >> 5;
  int lane = threadIdx.x & 31;
  int r  = lane & 15;
  int h8 = (lane >> 4) * 8;
  int mb = (lane >> 4) * 8;
  int cnt = counts[b];
  const bf16* Qh = Qb + (size_t)bh * NP * DH;
  const bf16* Kh = Kb + (size_t)bh * NP * DH;
  const bf16* Vh = VT + (size_t)bh * DH * NP;
  const float* bh_bias = bias + (size_t)bh * N * N;
  const float sc = 0.17677669529663687f;   // 1/sqrt(32)

#if HAVE_TDM
  if (w == 0) {
    tdm_load_1d((unsigned int)(size_t)&ldsK[0], Kh, NP * DH * 2);
    tdm_load_1d((unsigned int)(size_t)&ldsV[0], Vh, DH * NP * 2);
    __builtin_amdgcn_s_wait_tensorcnt(0);
  }
#else
  for (int i = threadIdx.x; i < NP * DH / 8; i += 256) {
    reinterpret_cast<v8bf*>(ldsK)[i] = reinterpret_cast<const v8bf*>(Kh)[i];
    reinterpret_cast<v8bf*>(ldsV)[i] = reinterpret_cast<const v8bf*>(Vh)[i];
  }
#endif
  __syncthreads();

  for (int qt = w; qt < NP / 16; qt += 8) {
    int q0 = qt * 16;
    v16bf aQ = ldfrag(Qh + (size_t)(q0 + r) * DH + h8);
    v8f acc0 = {}, acc1 = {};
    float mrow[8], lrow[8];
#pragma unroll
    for (int i = 0; i < 8; ++i) { mrow[i] = -1e30f; lrow[i] = 0.f; }

    for (int k0 = 0; k0 < NP; k0 += 32) {
      v16bf bK0 = ldfrag(ldsK + (size_t)(k0 + r)      * DH + h8);
      v16bf bK1 = ldfrag(ldsK + (size_t)(k0 + 16 + r) * DH + h8);
      v8f s0 = {}, s1 = {};
      s0 = wmma_bf16(aQ, bK0, s0);
      s1 = wmma_bf16(aQ, bK1, s1);

      float p0[8], p1[8];
#pragma unroll
      for (int i = 0; i < 8; ++i) {
        int qrow = q0 + mb + i;
        int c0 = k0 + r, c1 = k0 + 16 + r;
        float bia0 = (qrow < N && c0 < N) ? bh_bias[(size_t)qrow * N + c0] : 0.f;
        float bia1 = (qrow < N && c1 < N) ? bh_bias[(size_t)qrow * N + c1] : 0.f;
        p0[i] = (c0 < cnt) ? (s0[i] * sc + bia0) : -1e9f;
        p1[i] = (c1 < cnt) ? (s1[i] * sc + bia1) : -1e9f;
      }
      // online softmax per row (row = (half,vgpr); cols live in 16 lanes)
#pragma unroll
      for (int i = 0; i < 8; ++i) {
        float rm = fmaxf(p0[i], p1[i]);
#pragma unroll
        for (int off = 1; off < 16; off <<= 1)
          rm = fmaxf(rm, __shfl_xor(rm, off, 32));
        float mnew = fmaxf(mrow[i], rm);
        float corr = __expf(mrow[i] - mnew);
        mrow[i] = mnew;
        acc0[i] *= corr; acc1[i] *= corr;
        float e0 = __expf(p0[i] - mnew);
        float e1 = __expf(p1[i] - mnew);
        p0[i] = e0; p1[i] = e1;
        float rs = e0 + e1;
#pragma unroll
        for (int off = 1; off < 16; off <<= 1)
          rs += __shfl_xor(rs, off, 32);
        lrow[i] = lrow[i] * corr + rs;
      }
      // transpose P (C-layout) -> A-layout through per-wave LDS
#pragma unroll
      for (int i = 0; i < 8; ++i) {
        ldsP[w][(mb + i) * 32 + r]      = (bf16)p0[i];
        ldsP[w][(mb + i) * 32 + 16 + r] = (bf16)p1[i];
      }
      asm volatile("s_wait_dscnt 0" ::: "memory");
      v16bf aP  = ldfrag(&ldsP[w][r * 32 + h8]);
      v16bf bV0 = ldfrag(ldsV + (size_t)r        * NP + k0 + h8);
      v16bf bV1 = ldfrag(ldsV + (size_t)(16 + r) * NP + k0 + h8);
      acc0 = wmma_bf16(aP, bV0, acc0);
      acc1 = wmma_bf16(aP, bV1, acc1);
    }
    // normalize and store ctx into dense [B, NP, E] (head-major inside E)
#pragma unroll
    for (int i = 0; i < 8; ++i) {
      float inv = 1.0f / lrow[i];
      int qrow = q0 + mb + i;
      size_t base = ((size_t)b * NP + qrow) * E + hh * DH;
      ctxD[base + r]      = (bf16)(acc0[i] * inv);
      ctxD[base + 16 + r] = (bf16)(acc1[i] * inv);
    }
  }
}

// ---------------------------------------------------------------------------
// BatchNorm (training stats over all T tokens) + final residual
// ---------------------------------------------------------------------------
__global__ void k_colstats(const float* u, float* sum, float* sq, int T) {
  int c = threadIdx.x;
  int r0 = blockIdx.x * 64;
  float s = 0.f, q = 0.f;
  for (int r = 0; r < 64; ++r) {
    int t = r0 + r;
    if (t < T) { float v = u[(size_t)t * E + c]; s += v; q += v * v; }
  }
  atomicAdd(&sum[c], s);
  atomicAdd(&sq[c], q);
}

__global__ void k_bnfin(const float* sum, const float* sq, const float* g,
                        const float* be, float* scale, float* shift, float invT) {
  int c = threadIdx.x;
  float mean = sum[c] * invT;
  float var  = sq[c] * invT - mean * mean;
  float inv  = rsqrtf(var + 1e-5f);
  float s = g[c] * inv;
  scale[c] = s;
  shift[c] = be[c] - mean * s;
}

__global__ void k_bnapply(const float* y, const float* sc, const float* sh,
                          bf16* out, long n) {
  long i = blockIdx.x * (long)blockDim.x + threadIdx.x;
  if (i < n) { int c = (int)(i & (E - 1)); out[i] = (bf16)(y[i] * sc[c] + sh[c]); }
}

__global__ void k_final(const float* h2, const float* sc, const float* sh,
                        const float* y, float* out, long n) {
  long i = blockIdx.x * (long)blockDim.x + threadIdx.x;
  if (i < n) { int c = (int)(i & (E - 1)); out[i] = h2[i] * sc[c] + sh[c] + y[i]; }
}

// ---------------------------------------------------------------------------
extern "C" void kernel_launch(void* const* d_in, const int* in_sizes, int n_in,
                              void* d_out, int out_size, void* d_ws, size_t ws_size,
                              hipStream_t stream) {
  const float* x     = (const float*)d_in[0];
  const int*   bids  = (const int*)  d_in[1];
  const float* bias  = (const float*)d_in[2];
  const float* ln_g  = (const float*)d_in[3];
  const float* ln_b  = (const float*)d_in[4];
  const float* w_qkv = (const float*)d_in[5];
  const float* b_qkv = (const float*)d_in[6];
  const float* w_o   = (const float*)d_in[7];
  const float* b_o   = (const float*)d_in[8];
  const float* bn1_g = (const float*)d_in[9];
  const float* bn1_b = (const float*)d_in[10];
  const float* w1    = (const float*)d_in[11];
  const float* b1    = (const float*)d_in[12];
  const float* w2    = (const float*)d_in[13];
  const float* b2    = (const float*)d_in[14];
  const float* bn2_g = (const float*)d_in[15];
  const float* bn2_b = (const float*)d_in[16];

  const int T = in_sizes[1];
  int N = (int)lround(sqrt((double)(in_sizes[2] / (BB * HH))));   // 504

  // workspace partition (256B aligned)
  char* wsb = (char*)d_ws;
  size_t off = 0;
  auto alloc = [&](size_t bytes) -> void* {
    void* p = wsb + off;
    off = (off + bytes + 255) & ~(size_t)255;
    return p;
  };
  int*   counts = (int*)  alloc(BB * 4);
  int*   starts = (int*)  alloc(BB * 4);
  float* sums   = (float*)alloc(4 * E * 4);          // sum1, sq1, sum2, sq2
  float* scsh   = (float*)alloc(4 * E * 4);          // sc1, sh1, sc2, sh2
  bf16*  denseX = (bf16*) alloc((size_t)MP * E * 2);
  bf16*  Wqkvb  = (bf16*) alloc((size_t)3 * E * E * 2);
  bf16*  Wob    = (bf16*) alloc((size_t)E * E * 2);
  bf16*  W1b    = (bf16*) alloc((size_t)E * E * 2);
  bf16*  W2b    = (bf16*) alloc((size_t)E * E * 2);
  bf16*  Qb     = (bf16*) alloc((size_t)BB * HH * NP * DH * 2);
  bf16*  Kb     = (bf16*) alloc((size_t)BB * HH * NP * DH * 2);
  bf16*  VT     = (bf16*) alloc((size_t)BB * HH * NP * DH * 2);
  bf16*  ctxD   = (bf16*) alloc((size_t)MP * E * 2);
  float* y      = (float*)alloc((size_t)T * E * 4);
  bf16*  ybn    = (bf16*) alloc((size_t)T * E * 2);
  bf16*  h1     = (bf16*) alloc((size_t)T * E * 2);
  float* h2     = (float*)alloc((size_t)T * E * 4);
  (void)ws_size; (void)n_in; (void)out_size;

  float* sum1 = sums;         float* sq1 = sums + E;
  float* sum2 = sums + 2 * E; float* sq2 = sums + 3 * E;
  float* sc1 = scsh;          float* sh1 = scsh + E;
  float* sc2 = scsh + 2 * E;  float* sh2 = scsh + 3 * E;

  // 0) zero accumulators + padded dense input
  k_zero<<<1, 256, 0, stream>>>((unsigned int*)counts, BB);
  k_zero<<<(4 * E + 255) / 256, 256, 0, stream>>>((unsigned int*)sums, 4 * E);
  {
    long nwords = (long)MP * E / 2;     // bf16 pairs
    k_zero<<<(int)((nwords + 255) / 256), 256, 0, stream>>>((unsigned int*)denseX, nwords);
  }
  // 1) per-graph starts + counts
  k_starts<<<(T + 255) / 256, 256, 0, stream>>>(bids, starts, counts, T);
  // 2) weights -> bf16
  k_f2bf<<<(3 * E * E + 255) / 256, 256, 0, stream>>>(w_qkv, Wqkvb, 3L * E * E);
  k_f2bf<<<(E * E + 255) / 256, 256, 0, stream>>>(w_o, Wob, (long)E * E);
  k_f2bf<<<(E * E + 255) / 256, 256, 0, stream>>>(w1, W1b, (long)E * E);
  k_f2bf<<<(E * E + 255) / 256, 256, 0, stream>>>(w2, W2b, (long)E * E);
  // 3) LayerNorm + scatter to dense
  k_ln_scatter<<<T, 256, 0, stream>>>(x, bids, starts, ln_g, ln_b, denseX);
  // 4) QKV projection (WMMA), epilogue scatters Q/K/V^T per head
  k_gemm_qkv<<<dim3(3 * E / 128, MP / 64), 256, 0, stream>>>(denseX, Wqkvb, b_qkv,
                                                             Qb, Kb, VT);
  // 5) fused flash attention (bias streamed once; K/V staged via TDM)
  k_flash<<<BB * HH, 256, 0, stream>>>(Qb, Kb, VT, bias, counts, ctxD, N);
  // 6) output projection + gather + residual
  k_gemm_oproj<<<dim3(E / 128, MP / 64), 256, 0, stream>>>(ctxD, Wob, b_o, x,
                                                           starts, counts, y);
  // 7) BN1 stats -> scale/shift -> apply
  k_colstats<<<(T + 63) / 64, 256, 0, stream>>>(y, sum1, sq1, T);
  k_bnfin<<<1, 256, 0, stream>>>(sum1, sq1, bn1_g, bn1_b, sc1, sh1, 1.0f / (float)T);
  k_bnapply<<<(int)(((long)T * E + 255) / 256), 256, 0, stream>>>(y, sc1, sh1, ybn,
                                                                  (long)T * E);
  // 8) MLP: Linear+GELU (WMMA), Linear (WMMA)
  k_gemm_gelu<<<dim3(E / 128, T / 64), 256, 0, stream>>>(ybn, W1b, b1, h1);
  k_gemm_f32<<<dim3(E / 128, T / 64), 256, 0, stream>>>(h1, W2b, b2, h2);
  // 9) BN2 stats -> apply + final residual
  k_colstats<<<(T + 63) / 64, 256, 0, stream>>>(h2, sum2, sq2, T);
  k_bnfin<<<1, 256, 0, stream>>>(sum2, sq2, bn2_g, bn2_b, sc2, sh2, 1.0f / (float)T);
  k_final<<<(int)(((long)T * E + 255) / 256), 256, 0, stream>>>(h2, sc2, sh2, y,
                                                                (float*)d_out,
                                                                (long)T * E);
}
